// GainRNN_70540542869836
// MI455X (gfx1250) — matrix operand order repaired and checked by reading, not compiled, not run on hardware
//
#include <hip/hip_runtime.h>
#include <hip/hip_fp16.h>

typedef _Float16 f16;
typedef __attribute__((ext_vector_type(16))) _Float16 v16h;
typedef __attribute__((ext_vector_type(8)))  _Float16 v8h;
typedef __attribute__((ext_vector_type(8)))  float    v8f;

#define B_   64
#define T_   512
#define DIN  256
#define H_   1024
#define DOUT 64
#define INVH (1.0f / 1024.0f)

// ---------------- workspace layout (all sizes 256B-aligned) ----------------
constexpr size_t SZ_WI  = (size_t)H_ * DIN * 2;     // Wi   f16 [H][DIN]
constexpr size_t SZ_J   = (size_t)H_ * H_ * 2;      // J    f16 [H][H]
constexpr size_t SZ_WO  = (size_t)DOUT * H_ * 2;    // Wout f16 [O][H]
constexpr size_t SZ_XH  = (size_t)B_ * T_ * DIN * 2;// x    f16 [B][T][DIN]
constexpr size_t SZ_XIN = (size_t)T_ * B_ * H_ * 2; // x_in f16 [T][B][H]
constexpr size_t OFF_WI  = 0;
constexpr size_t OFF_J   = OFF_WI + SZ_WI;
constexpr size_t OFF_WO  = OFF_J + SZ_J;
constexpr size_t OFF_XH  = OFF_WO + SZ_WO;
constexpr size_t OFF_XIN = OFF_XH + SZ_XH;
constexpr size_t OFF_HS  = OFF_XIN + SZ_XIN;        // hs  f16 [T][B][H]

// ---------------- prep: W = bg + proj @ recv^T (rank 2), cast f16 ----------
__global__ void build_lowrank_f16(const float* __restrict__ bg,
                                  const float* __restrict__ proj,
                                  const float* __restrict__ recv,
                                  f16* __restrict__ out, int rows, int cols) {
    int idx = blockIdx.x * blockDim.x + threadIdx.x;
    if (idx >= rows * cols) return;
    int r = idx / cols, c = idx % cols;
    float v = bg[idx] + proj[r * 2 + 0] * recv[c * 2 + 0]
                      + proj[r * 2 + 1] * recv[c * 2 + 1];
    out[idx] = (f16)v;
}

__global__ void cvt_f16(const float* __restrict__ in, f16* __restrict__ out, int n) {
    int idx = blockIdx.x * blockDim.x + threadIdx.x;
    if (idx < n) out[idx] = (f16)in[idx];
}

// helper: assemble v16h from two v8h chunks
__device__ __forceinline__ v16h mk16(v8h lo, v8h hi) {
    v16h r;
#pragma unroll
    for (int i = 0; i < 8; ++i) { r[i] = lo[i]; r[i + 8] = hi[i]; }
    return r;
}

// ---------------- x_in[t][b][h] = Wi @ x  (one wave = one 16x16 tile) ------
__global__ void xin_gemm(const f16* __restrict__ xh, const f16* __restrict__ Wi,
                         f16* __restrict__ xin) {
    int wave = (blockIdx.x * blockDim.x + threadIdx.x) >> 5;
    int lane = threadIdx.x & 31;
    // tiles: b(64) x t0(32) x h0(64)
    int h0 = (wave & 63) * 16;
    int t0 = ((wave >> 6) & 31) * 16;
    int b  = wave >> 11;
    int lm     = lane & 15;
    int kA     = (lane >> 4) * 8;   // A-fragment K sub-offset
    int kB     = (lane >> 4) * 8;   // B-fragment K sub-offset (same split as A)
    const f16* arow = xh + ((size_t)b * T_ + t0 + lm) * DIN;
    const f16* brow = Wi + (size_t)(h0 + lm) * DIN;   // B col n = Wi row n
    v8f c = {};
#pragma unroll
    for (int kt = 0; kt < DIN / 32; ++kt) {
        int ka = kt * 32 + kA;
        int kb = kt * 32 + kB;
        v16h a = mk16(*(const v8h*)(arow + ka), *(const v8h*)(arow + ka + 16));
        v16h bm = mk16(*(const v8h*)(brow + kb), *(const v8h*)(brow + kb + 16));
        c = __builtin_amdgcn_wmma_f32_16x16x32_f16(false, a, false, bm,
                                                   (short)0, c, false, false);
    }
    int n = h0 + lm;
    int mb = (lane >> 4) * 8;
#pragma unroll
    for (int i = 0; i < 8; ++i) {
        int t = t0 + mb + i;
        xin[((size_t)t * B_ + b) * H_ + n] = (f16)c[i];
    }
}

// ---------------- persistent scan: 4 WGs x 16 batches, no global sync ------
__global__ void __launch_bounds__(1024)
scan_kernel(const f16* __restrict__ J, const f16* __restrict__ xin,
            const float* __restrict__ gain, const float* __restrict__ thr,
            f16* __restrict__ hs) {
    extern __shared__ f16 smem[];
    f16* hbuf = smem;             // [16][1024] current h (f16)
    f16* xbuf = smem + 16 * H_;   // [16][1024] staged x_in tile
    const int tid  = threadIdx.x;
    const int bg   = blockIdx.x;        // batch group 0..3 (16 batches each)
    const int wave = tid >> 5;          // 0..31, owns 32 h-columns
    const int lane = tid & 31;
    const int lm   = lane & 15;
    const int kA   = (lane >> 4) * 8;
    const int kB   = (lane >> 4) * 8;
    const int col0 = wave * 32;

    for (int i = tid; i < 16 * H_; i += 1024) hbuf[i] = (f16)0.0f;
    __syncthreads();

    const int n0 = col0 + lm, n1 = col0 + 16 + lm;
    const float g0 = gain[n0] * INVH, g1 = gain[n1] * INVH;
    const float th0 = thr[n0], th1 = thr[n1];
    const int mb = (lane >> 4) * 8;
    const f16* jrow0 = J + (size_t)n0 * H_;   // B col n = J row n (Jt GEMM)
    const f16* jrow1 = J + (size_t)n1 * H_;
    const f16* hrow  = hbuf + (size_t)lm * H_;

    for (int t = 0; t < T_; ++t) {
        // stage this step's x_in tile (32KB contiguous) into registers
        const f16* xg = xin + ((size_t)t * B_ + bg * 16) * H_;
        v8h x0 = *(const v8h*)(xg + tid * 16);
        v8h x1 = *(const v8h*)(xg + tid * 16 + 8);
        if (t + 1 < T_)
            __builtin_prefetch(xin + ((size_t)(t + 1) * B_ + bg * 16) * H_ + tid * 16, 0, 1);

        // r_in = h @ J^T : each wave produces [16 x 32] in fp32
        v8f c0 = {}, c1 = {};
#pragma unroll 4
        for (int kt = 0; kt < H_ / 32; ++kt) {
            int ka = kt * 32 + kA;
            int kb = kt * 32 + kB;
            v16h a  = mk16(*(const v8h*)(hrow + ka), *(const v8h*)(hrow + ka + 16));
            v16h b0 = mk16(*(const v8h*)(jrow0 + kb), *(const v8h*)(jrow0 + kb + 16));
            v16h b1 = mk16(*(const v8h*)(jrow1 + kb), *(const v8h*)(jrow1 + kb + 16));
            c0 = __builtin_amdgcn_wmma_f32_16x16x32_f16(false, a, false, b0,
                                                        (short)0, c0, false, false);
            c1 = __builtin_amdgcn_wmma_f32_16x16x32_f16(false, a, false, b1,
                                                        (short)0, c1, false, false);
        }

        // publish staged x_in to LDS
        *(v8h*)(xbuf + tid * 16)     = x0;
        *(v8h*)(xbuf + tid * 16 + 8) = x1;
        __syncthreads();   // all reads of hbuf done; xbuf visible

        // h_new = gain * relu(r_in + x_in - thr) / H   (overwrite hbuf in place)
#pragma unroll
        for (int i = 0; i < 8; ++i) {
            int m = mb + i;
            float p0 = c0[i] + (float)xbuf[m * H_ + n0] - th0;
            float p1 = c1[i] + (float)xbuf[m * H_ + n1] - th1;
            hbuf[m * H_ + n0] = (f16)(p0 > 0.0f ? p0 * g0 : 0.0f);
            hbuf[m * H_ + n1] = (f16)(p1 > 0.0f ? p1 * g1 : 0.0f);
        }
        __syncthreads();   // hbuf complete for everyone

        // coalesced store of hs[t][bg*16 .. +16][:] (32KB contiguous)
        f16* hg = hs + ((size_t)t * B_ + bg * 16) * H_;
        *(v8h*)(hg + tid * 16)     = *(const v8h*)(hbuf + tid * 16);
        *(v8h*)(hg + tid * 16 + 8) = *(const v8h*)(hbuf + tid * 16 + 8);
    }
}

// ---------------- out[b][t][o] = hs @ W^T + bias (one wave per 16x16) ------
__global__ void readout_gemm(const f16* __restrict__ hs, const f16* __restrict__ Wo,
                             const float* __restrict__ bias, float* __restrict__ out) {
    int wave = (blockIdx.x * blockDim.x + threadIdx.x) >> 5;
    int lane = threadIdx.x & 31;
    // tiles: t(512) x b0(4) x o0(4)
    int o0 = (wave & 3) * 16;
    int b0 = ((wave >> 2) & 3) * 16;
    int t  = wave >> 4;
    if (t >= T_) return;
    int lm = lane & 15;
    int kA = (lane >> 4) * 8;
    int kB = (lane >> 4) * 8;
    const f16* arow = hs + ((size_t)t * B_ + b0 + lm) * H_;
    const f16* brow = Wo + (size_t)(o0 + lm) * H_;
    v8f c = {};
#pragma unroll 4
    for (int kt = 0; kt < H_ / 32; ++kt) {
        int ka = kt * 32 + kA;
        int kb = kt * 32 + kB;
        v16h a  = mk16(*(const v8h*)(arow + ka), *(const v8h*)(arow + ka + 16));
        v16h bm = mk16(*(const v8h*)(brow + kb), *(const v8h*)(brow + kb + 16));
        c = __builtin_amdgcn_wmma_f32_16x16x32_f16(false, a, false, bm,
                                                   (short)0, c, false, false);
    }
    int n = o0 + lm;
    float bv = bias[n];
    int mb = (lane >> 4) * 8;
#pragma unroll
    for (int i = 0; i < 8; ++i) {
        int b = b0 + mb + i;
        out[((size_t)b * T_ + t) * DOUT + n] = c[i] + bv;
    }
}

// ---------------------------------------------------------------------------
extern "C" void kernel_launch(void* const* d_in, const int* in_sizes, int n_in,
                              void* d_out, int out_size, void* d_ws, size_t ws_size,
                              hipStream_t stream) {
    const float* x       = (const float*)d_in[0];
    const float* in_bg   = (const float*)d_in[1];
    const float* in_proj = (const float*)d_in[2];
    const float* in_recv = (const float*)d_in[3];
    const float* gain    = (const float*)d_in[4];
    const float* thr     = (const float*)d_in[5];
    const float* rec_bg  = (const float*)d_in[6];
    const float* rec_proj= (const float*)d_in[7];
    const float* rec_recv= (const float*)d_in[8];
    const float* w_out   = (const float*)d_in[9];
    const float* b_out   = (const float*)d_in[10];
    float* out = (float*)d_out;

    char* ws = (char*)d_ws;
    f16* Wi  = (f16*)(ws + OFF_WI);
    f16* Jm  = (f16*)(ws + OFF_J);
    f16* Wo  = (f16*)(ws + OFF_WO);
    f16* xh  = (f16*)(ws + OFF_XH);
    f16* xin = (f16*)(ws + OFF_XIN);
    f16* hsb = (f16*)(ws + OFF_HS);

    // prep: low-rank-corrected weights + f16 casts
    build_lowrank_f16<<<(H_ * DIN + 255) / 256, 256, 0, stream>>>(
        in_bg, in_proj, in_recv, Wi, H_, DIN);
    build_lowrank_f16<<<(H_ * H_ + 255) / 256, 256, 0, stream>>>(
        rec_bg, rec_proj, rec_recv, Jm, H_, H_);
    cvt_f16<<<(B_ * T_ * DIN + 255) / 256, 256, 0, stream>>>(x, xh, B_ * T_ * DIN);
    cvt_f16<<<(DOUT * H_ + 255) / 256, 256, 0, stream>>>(w_out, Wo, DOUT * H_);

    // x_in: (B*T/16)*(H/16) = 131072 tiles, 8 waves / 256-thread block
    xin_gemm<<<131072 / 8, 256, 0, stream>>>(xh, Wi, xin);

    // persistent scan: 4 WGs (one per 16-batch group), 32 waves each, 64KB LDS
    scan_kernel<<<4, 1024, 64 * 1024, stream>>>(Jm, xin, gain, thr, hsb);

    // readout: 512*4*4 = 8192 tiles, 8 waves / block
    readout_gemm<<<8192 / 8, 256, 0, stream>>>(hsb, Wo, b_out, out);
}
